// MultiHeadAttention_87170656240144
// MI455X (gfx1250) — matrix operand order repaired and checked by reading, not compiled, not run on hardware
//
#include <hip/hip_runtime.h>
#include <hip/hip_bf16.h>

// ---------------------------------------------------------------------------
// MHA on gfx1250 (MI455X): all matmuls via v_wmma_f32_16x16x32_bf16 (wave32).
//   B=32 T=512 D=1024 H=16 dk=64
// Pipeline:
//   cvt fp32->bf16 (x, w_q, w_k, w_v, w_o)
//   proj<0>: Q = (x@Wq^T)*0.125 -> bf16 [B,H,T,dk]  (async-LDS staged GEMM)
//   proj<0>: K = x@Wk^T  -> bf16 [B,H,T,dk]
//   proj<1>: V = x@Wv^T  -> bf16 [B,H,dk,T]         (transposed for PV B-frags)
//   attn   : flash-style causal softmax; unmasked main loop + masked diagonal
//   proj<2>: out = ctx@Wo^T + b_o -> fp32
// ---------------------------------------------------------------------------

#define BN 32
#define TN 512
#define DN 1024
#define HN 16
#define DKN 64

typedef __attribute__((ext_vector_type(16))) __bf16 v16bf;
typedef __attribute__((ext_vector_type(8)))  float  v8f;
typedef __attribute__((address_space(3))) unsigned short lds_ushort;

template <bool B> struct BoolC { static constexpr bool value = B; };

static __device__ __forceinline__ unsigned short f2bf(float f) {
    unsigned u = __float_as_uint(f);
    unsigned r = u + 0x7FFFu + ((u >> 16) & 1u);   // round to nearest even
    return (unsigned short)(r >> 16);
}

static __device__ __forceinline__ v8f wmma_bf16(v16bf a, v16bf b, v8f c) {
    return __builtin_amdgcn_wmma_f32_16x16x32_bf16(
        /*neg_a=*/false, a, /*neg_b=*/false, b,
        /*c_mod=*/(short)0, c, /*reuse_a=*/false, /*reuse_b=*/false);
}

// Per-lane async copy of 16B global -> LDS (GLOBAL_LOAD_ASYNC_TO_LDS_B128,
// tracked by ASYNCcnt).  VDST operand is the 32-bit LDS address.
static __device__ __forceinline__ void async_copy_b128(const unsigned short* g,
                                                       unsigned short* l) {
    asm volatile("global_load_async_to_lds_b128 %0, %1, off"
                 :: "v"((lds_ushort*)l), "v"(g)
                 : "memory");
}

static __device__ __forceinline__ void wait_async0() {
#if __has_builtin(__builtin_amdgcn_s_wait_asynccnt)
    __builtin_amdgcn_s_wait_asynccnt(0);
#else
    asm volatile("s_wait_asynccnt 0" ::: "memory");
#endif
}

// A-fragment (16x32 bf16): lane r = row; lanes 0-15 hold k=0..15,
// lanes 16-31 hold k=16..31 (same k order as ld_frag_B)
static __device__ __forceinline__ v16bf ld_frag_A(const unsigned short* row, int hi) {
    union { uint4 u[2]; v16bf v; } f;
    f.u[0] = *(const uint4*)(row + hi * 16);
    f.u[1] = *(const uint4*)(row + hi * 16 + 8);
    return f.v;
}

// B-fragment (32x16 bf16): lane r = column (a row of the K-major source);
// lanes 0-15 hold k=0..15, lanes 16-31 hold k=16..31
static __device__ __forceinline__ v16bf ld_frag_B(const unsigned short* row, int hi) {
    union { uint4 u[2]; v16bf v; } f;
    const unsigned short* p = row + hi * 16;
    f.u[0] = *(const uint4*)(p);
    f.u[1] = *(const uint4*)(p + 8);
    return f.v;
}

// ---------------------------------------------------------------------------
__global__ __launch_bounds__(256) void cvt_bf16_kernel(const float* __restrict__ src,
                                                       unsigned short* __restrict__ dst,
                                                       int n4) {
    int i = blockIdx.x * blockDim.x + threadIdx.x;
    if (i < n4) {
        float4 f = ((const float4*)src)[i];
        uint2 o;
        o.x = (unsigned)f2bf(f.x) | ((unsigned)f2bf(f.y) << 16);
        o.y = (unsigned)f2bf(f.z) | ((unsigned)f2bf(f.w) << 16);
        ((uint2*)dst)[i] = o;
    }
}

// ---------------------------------------------------------------------------
// GEMM: C[M=16384, N=1024] = oscale * (A[M,1024] @ W[N,1024]^T)
// Block tile 128(M) x 128(N); 8 waves, each 32x64 (2x4 accumulators).
// K-step 32, double-buffered async global->LDS staging shared by all waves.
// LDS row stride 40 ushorts (80B) -> bank-balanced ds_load_b128 reads.
// MODE 0: store bf16 [B,H,T,dk]   (Q, K)
// MODE 1: store bf16 [B,H,dk,T]   (V transposed)
// MODE 2: store f32  [M,N] + bias (output projection)
template <int MODE>
__global__ __launch_bounds__(256) void proj_kernel(const unsigned short* __restrict__ A,
                                                   const unsigned short* __restrict__ W,
                                                   const float* __restrict__ bias,
                                                   unsigned short* __restrict__ out16,
                                                   float* __restrict__ out32,
                                                   float oscale) {
    __shared__ unsigned short Abuf[2][128 * 40];
    __shared__ unsigned short Wbuf[2][128 * 40];

    const int tid  = threadIdx.x;
    const int w    = tid >> 5;
    const int lane = tid & 31;
    const int r    = lane & 15;
    const int hi   = lane >> 4;

    const int m_base = blockIdx.x * 128;
    const int n_base = blockIdx.y * 128;
    const int wm = (w >> 1) * 32;        // wave M offset: 0/32/64/96
    const int wn = (w & 1) * 64;         // wave N offset: 0/64

    // cooperative loader assignment: 2 threads per row, 16 elems each
    const int lrow  = tid >> 1;          // 0..127
    const int lhalf = tid & 1;
    const unsigned short* gA = A + (size_t)(m_base + lrow) * DN + lhalf * 16;
    const unsigned short* gW = W + (size_t)(n_base + lrow) * DN + lhalf * 16;
    unsigned short* lA[2] = { &Abuf[0][lrow * 40 + lhalf * 16],
                              &Abuf[1][lrow * 40 + lhalf * 16] };
    unsigned short* lW[2] = { &Wbuf[0][lrow * 40 + lhalf * 16],
                              &Wbuf[1][lrow * 40 + lhalf * 16] };

    v8f acc[2][4] = {};

    // prologue: stage k-step 0 into buffer 0
    async_copy_b128(gA + 0, lA[0]);
    async_copy_b128(gA + 8, lA[0] + 8);
    async_copy_b128(gW + 0, lW[0]);
    async_copy_b128(gW + 8, lW[0] + 8);

    int cur = 0;
    for (int k0 = 0; k0 < DN; k0 += 32) {
        wait_async0();          // this wave's writes for buf[cur] landed
        __syncthreads();        // everyone's writes landed; prev reads done

        if (k0 + 32 < DN) {     // stage next k-step into the other buffer
            const int kn = k0 + 32;
            unsigned short* dA = lA[cur ^ 1];
            unsigned short* dW = lW[cur ^ 1];
            async_copy_b128(gA + kn,     dA);
            async_copy_b128(gA + kn + 8, dA + 8);
            async_copy_b128(gW + kn,     dW);
            async_copy_b128(gW + kn + 8, dW + 8);
        }

        const unsigned short* Ab = &Abuf[cur][0];
        const unsigned short* Wb = &Wbuf[cur][0];
        v16bf af0 = ld_frag_A(Ab + (wm + r) * 40, hi);
        v16bf af1 = ld_frag_A(Ab + (wm + 16 + r) * 40, hi);
#pragma unroll
        for (int nb = 0; nb < 4; ++nb) {
            v16bf bf = ld_frag_B(Wb + (wn + nb * 16 + r) * 40, hi);
            acc[0][nb] = wmma_bf16(af0, bf, acc[0][nb]);
            acc[1][nb] = wmma_bf16(af1, bf, acc[1][nb]);
        }
        cur ^= 1;
    }

#pragma unroll
    for (int mb = 0; mb < 2; ++mb) {
#pragma unroll
        for (int nb = 0; nb < 4; ++nb) {
#pragma unroll
            for (int v = 0; v < 8; ++v) {
                const float val = acc[mb][nb][v] * oscale;
                const int m = m_base + wm + mb * 16 + v + 8 * hi;  // = b*T + t
                const int n = n_base + wn + nb * 16 + r;           // = h*dk + d
                if (MODE == 2) {
                    out32[(size_t)m * DN + n] = val + bias[n];
                } else {
                    const int b = m >> 9, t = m & 511;
                    const int h = n >> 6, d = n & 63;
                    if (MODE == 0)
                        out16[(((size_t)(b * HN + h) * TN) + t) * DKN + d] = f2bf(val);
                    else
                        out16[(((size_t)(b * HN + h) * DKN) + d) * TN + t] = f2bf(val);
                }
            }
        }
    }
}

// ---------------------------------------------------------------------------
// Flash attention (causal). Block = 8 waves; wave owns 16 query rows.
// Q is pre-scaled by 1/sqrt(dk) at projection time.
// Interior tiles run with zero masking VALU; only the diagonal tile masks.
__global__ __launch_bounds__(256) void attn_kernel(const unsigned short* __restrict__ q,
                                                   const unsigned short* __restrict__ k,
                                                   const unsigned short* __restrict__ vt,
                                                   unsigned short* __restrict__ ctx) {
    __shared__ unsigned short p_lds[8][16 * 40];   // per-wave 16x32 bf16, padded rows

    const int tid  = threadIdx.x;
    const int w    = tid >> 5;
    const int lane = tid & 31;
    const int r    = lane & 15;
    const int hi   = lane >> 4;

    const int b  = blockIdx.z;
    const int h  = blockIdx.y;
    const int q0 = blockIdx.x * 128 + w * 16;

    const unsigned short* qbase = q  + (size_t)(b * HN + h) * TN * DKN;
    const unsigned short* kbase = k  + (size_t)(b * HN + h) * TN * DKN;
    const unsigned short* vbase = vt + (size_t)(b * HN + h) * DKN * TN;

    v16bf qf0 = ld_frag_A(qbase + (size_t)(q0 + r) * DKN + 0,  hi);
    v16bf qf1 = ld_frag_A(qbase + (size_t)(q0 + r) * DKN + 32, hi);

    v8f acc0 = {}, acc1 = {}, acc2 = {}, acc3 = {};
    float mrow[8], lrow[8];
#pragma unroll
    for (int v = 0; v < 8; ++v) { mrow[v] = -3.0e38f; lrow[v] = 0.0f; }

    unsigned short* pw = &p_lds[w][0];

    auto do_tile = [&](int j0, auto masked_c) {
        constexpr bool MASKED = decltype(masked_c)::value;

        if (j0 + 32 < q0 + 16) {                   // prefetch next K tile rows
            __builtin_prefetch(kbase + (size_t)(j0 + 32 + r) * DKN, 0, 0);
            __builtin_prefetch(kbase + (size_t)(j0 + 48 + r) * DKN, 0, 0);
        }

        // ---- S = Q @ K^T (two 16-key column blocks) ----
        v8f s0 = {}, s1 = {};
        {
            v16bf kf;
            kf = ld_frag_B(kbase + (size_t)(j0 + r) * DKN + 0,  hi); s0 = wmma_bf16(qf0, kf, s0);
            kf = ld_frag_B(kbase + (size_t)(j0 + r) * DKN + 32, hi); s0 = wmma_bf16(qf1, kf, s0);
            kf = ld_frag_B(kbase + (size_t)(j0 + 16 + r) * DKN + 0,  hi); s1 = wmma_bf16(qf0, kf, s1);
            kf = ld_frag_B(kbase + (size_t)(j0 + 16 + r) * DKN + 32, hi); s1 = wmma_bf16(qf1, kf, s1);
        }

        // ---- online softmax; rows span 16 lanes (D-frag layout) ----
#pragma unroll
        for (int v = 0; v < 8; ++v) {
            float a0 = s0[v];
            float a1 = s1[v];
            if (MASKED) {
                const int row = q0 + v + 8 * hi;
                if (j0 + r      > row) a0 = -3.0e38f;
                if (j0 + 16 + r > row) a1 = -3.0e38f;
            }

            float rm = fmaxf(a0, a1);
#pragma unroll
            for (int o = 1; o < 16; o <<= 1) rm = fmaxf(rm, __shfl_xor(rm, o, 16));
            const float mn = fmaxf(mrow[v], rm);

            const float p0 = __expf(a0 - mn);
            const float p1 = __expf(a1 - mn);
            float rs = p0 + p1;
#pragma unroll
            for (int o = 1; o < 16; o <<= 1) rs += __shfl_xor(rs, o, 16);

            const float alpha = __expf(mrow[v] - mn);
            mrow[v] = mn;
            lrow[v] = lrow[v] * alpha + rs;
            acc0[v] *= alpha; acc1[v] *= alpha; acc2[v] *= alpha; acc3[v] *= alpha;

            pw[(v + 8 * hi) * 40 + r]      = f2bf(p0);
            pw[(v + 8 * hi) * 40 + 16 + r] = f2bf(p1);
        }

        // ---- O += P @ V  (P reloaded in A-layout from LDS; same-wave DS in-order) ----
        v16bf pf = ld_frag_A(pw + r * 40, hi);
        v16bf vf;
        vf = ld_frag_B(vbase + (size_t)(0 * 16 + r) * TN + j0, hi); acc0 = wmma_bf16(pf, vf, acc0);
        vf = ld_frag_B(vbase + (size_t)(1 * 16 + r) * TN + j0, hi); acc1 = wmma_bf16(pf, vf, acc1);
        vf = ld_frag_B(vbase + (size_t)(2 * 16 + r) * TN + j0, hi); acc2 = wmma_bf16(pf, vf, acc2);
        vf = ld_frag_B(vbase + (size_t)(3 * 16 + r) * TN + j0, hi); acc3 = wmma_bf16(pf, vf, acc3);
    };

    // tiles with j0+31 <= q0 are fully below the diagonal for all 16 rows
    const int jfull_end = (q0 >= 31) ? ((((q0 - 31) >> 5) << 5) + 32) : 0;

    int j0 = 0;
    for (; j0 < jfull_end; j0 += 32) do_tile(j0, BoolC<false>{});
    for (; j0 < q0 + 16;   j0 += 32) do_tile(j0, BoolC<true>{});

    // ---- epilogue: O /= l, store ctx bf16 [B,T,H,dk] ----
#pragma unroll
    for (int v = 0; v < 8; ++v) {
        const float rinv = 1.0f / lrow[v];
        const int t = q0 + v + 8 * hi;
        const size_t base = ((size_t)(b * TN + t) * HN + h) * DKN;
        ctx[base + 0  + r] = f2bf(acc0[v] * rinv);
        ctx[base + 16 + r] = f2bf(acc1[v] * rinv);
        ctx[base + 32 + r] = f2bf(acc2[v] * rinv);
        ctx[base + 48 + r] = f2bf(acc3[v] * rinv);
    }
}

// ---------------------------------------------------------------------------
extern "C" void kernel_launch(void* const* d_in, const int* in_sizes, int n_in,
                              void* d_out, int out_size, void* d_ws, size_t ws_size,
                              hipStream_t stream) {
    (void)in_sizes; (void)n_in; (void)out_size; (void)ws_size;

    const float* x  = (const float*)d_in[0];
    const float* wq = (const float*)d_in[1];
    const float* wk = (const float*)d_in[2];
    const float* wv = (const float*)d_in[3];
    const float* wo = (const float*)d_in[4];
    const float* bo = (const float*)d_in[5];
    float* out = (float*)d_out;

    const size_t XN = (size_t)BN * TN * DN;   // 16,777,216
    const size_t WN = (size_t)DN * DN;        //  1,048,576

    unsigned short* base = (unsigned short*)d_ws;
    unsigned short* xb  = base;
    unsigned short* wqb = xb  + XN;
    unsigned short* wkb = wqb + WN;
    unsigned short* wvb = wkb + WN;
    unsigned short* wob = wvb + WN;
    unsigned short* qws = wob + WN;
    unsigned short* kws = qws + XN;
    unsigned short* vws = kws + XN;   // transposed [B,H,dk,T]
    unsigned short* cws = vws + XN;   // ctx [B,T,H,dk]

    // fp32 -> bf16 conversions
    cvt_bf16_kernel<<<(int)(XN / 4 / 256), 256, 0, stream>>>(x,  xb,  (int)(XN / 4));
    cvt_bf16_kernel<<<(int)(WN / 4 / 256), 256, 0, stream>>>(wq, wqb, (int)(WN / 4));
    cvt_bf16_kernel<<<(int)(WN / 4 / 256), 256, 0, stream>>>(wk, wkb, (int)(WN / 4));
    cvt_bf16_kernel<<<(int)(WN / 4 / 256), 256, 0, stream>>>(wv, wvb, (int)(WN / 4));
    cvt_bf16_kernel<<<(int)(WN / 4 / 256), 256, 0, stream>>>(wo, wob, (int)(WN / 4));

    // projections: block tile 128x128 -> grid (16384/128, 1024/128)
    dim3 pgrid(128, 8);
    proj_kernel<0><<<pgrid, 256, 0, stream>>>(xb, wqb, nullptr, qws, nullptr, 0.125f);
    proj_kernel<0><<<pgrid, 256, 0, stream>>>(xb, wkb, nullptr, kws, nullptr, 1.0f);
    proj_kernel<1><<<pgrid, 256, 0, stream>>>(xb, wvb, nullptr, vws, nullptr, 1.0f);

    // attention: grid = (T/128, H, B)
    attn_kernel<<<dim3(TN / 128, HN, BN), 256, 0, stream>>>(qws, kws, vws, cws);

    // output projection + bias -> fp32
    proj_kernel<2><<<pgrid, 256, 0, stream>>>(cws, wob, bo, nullptr, out, 1.0f);
}
